// OpusBlock_34445637714413
// MI455X (gfx1250) — matrix operand order repaired and checked by reading, not compiled, not run on hardware
//
#include <hip/hip_runtime.h>
#include <hip/hip_bf16.h>
#include <stdint.h>

// ---------------------------------------------------------------------------
// MI455X transformer block: RMSNorm -> QKV proj -> RoPE -> flash attention
// (ALiBi + causal) -> O proj + residual -> RMSNorm -> SwiGLU FFN + residual.
// Matmuls: v_wmma_f32_16x16x32_bf16 (fp32 accumulate).
// GEMM tile staging: TDM tensor_load_to_lds (6-arg clang-23 builtin) with
// native LDS row padding; fallback = global_load_b128 + ds_store_b128.
// ---------------------------------------------------------------------------

typedef __attribute__((ext_vector_type(16))) __bf16 v16bf;
typedef __attribute__((ext_vector_type(8)))  float  v8f;
typedef __attribute__((ext_vector_type(4)))  unsigned int v4u;
typedef __attribute__((ext_vector_type(8)))  int    v8i;
typedef __attribute__((ext_vector_type(4)))  int    v4i;

#define CB 2
#define CS 2048
#define CD 2048
#define CH 16
#define CHD 128
#define CFF 8192
#define CM (CB*CS)   // 4096 rows

__device__ __forceinline__ uint16_t f2bf(float f) {
  uint32_t u = __builtin_bit_cast(uint32_t, f);
  u += 0x7FFFu + ((u >> 16) & 1u);            // round-to-nearest-even
  return (uint16_t)(u >> 16);
}
__device__ __forceinline__ float bf2f(uint16_t h) {
  uint32_t u = ((uint32_t)h) << 16;
  return __builtin_bit_cast(float, u);
}
__device__ __forceinline__ v8f vzero8() {
  v8f z;
  #pragma unroll
  for (int i = 0; i < 8; ++i) z[i] = 0.f;
  return z;
}

// A fragment (16x32, MxK): lane<16 -> row=lane, K {0..7}U{16..23};
// lane>=16 -> row=lane-16, K {8..15}U{24..31}. Two 16B LDS loads.
__device__ __forceinline__ v16bf frag_a(const uint16_t* tile, int pitch) {
  int lane = threadIdx.x & 31;
  const uint16_t* p = tile + (lane & 15) * pitch + ((lane >> 4) << 3);
  union { v16bf v; uint4 q[2]; } u;
  u.q[0] = *(const uint4*)(p);
  u.q[1] = *(const uint4*)(p + 16);
  return u.v;
}
// B fragment (32x16, KxN): lane%16 -> col n, K {0..7}U{16..23} (lane<16), {8..15}U{24..31} (lane>=16).
__device__ __forceinline__ v16bf frag_b(const uint16_t* tile, int pitch) {
  int lane = threadIdx.x & 31;
  const uint16_t* p = tile + (lane & 15) * pitch + ((lane >> 4) << 3);
  union { v16bf v; uint4 q[2]; } u;
  u.q[0] = *(const uint4*)(p);
  u.q[1] = *(const uint4*)(p + 16);
  return u.v;
}

// ---------------------------------------------------------------------------
// TDM: load a [128 rows x 32 bf16] tile (row stride = strideElems) from
// global into LDS at lds_off, padding each 64B row with 16B so the LDS
// pitch is 80B (= GP elems), matching the WMMA fragment addressing.
// D# layout per CDNA5 ISA.
// This toolchain exposes the 6-arg builtin:
//   (uint32x4 g0, int32x8 g1, int32x4 g2, int32x4 g3, int32x8 extra, i32 cpol)
// ---------------------------------------------------------------------------
#if __has_builtin(__builtin_amdgcn_tensor_load_to_lds) && \
    __has_builtin(__builtin_amdgcn_s_wait_tensorcnt)
#define USE_TDM 1
__device__ __forceinline__ void tdm_load_tile_128x32(
    const uint16_t* gbase, uint32_t lds_off, int rowsTotal, int kTotal,
    int strideElems) {
  uint64_t ga = (uint64_t)(uintptr_t)gbase;
  v4u g0;
  g0[0] = 1u;                                   // count=1, user descriptor
  g0[1] = lds_off;                              // lds_addr (bytes)
  g0[2] = (unsigned)(ga & 0xFFFFFFFFu);         // global_addr[31:0]
  g0[3] = (unsigned)((ga >> 32) & 0x01FFFFFFu)  // global_addr[56:32]
          | (2u << 30);                         // type = 2 ("image")
  v8i g1;
  // data_size=1 (2B) | pad_enable | pad_interval=3 (16 DW) | pad_amount=3 (4 DW)
  g1[0] = (1 << 16) | (1 << 20) | (3 << 22) | (3 << 25);
  g1[1] = (int)(((unsigned)kTotal & 0xFFFFu) << 16);        // tensor_dim0 lo16
  g1[2] = (int)((((unsigned)kTotal >> 16) & 0xFFFFu) |
                (((unsigned)rowsTotal & 0xFFFFu) << 16));   // dim0 hi | dim1 lo
  g1[3] = (int)((((unsigned)rowsTotal >> 16) & 0xFFFFu) |
                (32u << 16));                               // dim1 hi | tile_dim0=32
  g1[4] = 128;                                  // tile_dim1=128, tile_dim2=0
  g1[5] = strideElems;                          // tensor_dim0_stride lo32
  g1[6] = 0;
  g1[7] = 0;
  v4i z4; z4[0] = 0; z4[1] = 0; z4[2] = 0; z4[3] = 0;
  v8i z8;
  #pragma unroll
  for (int i = 0; i < 8; ++i) z8[i] = 0;
  __builtin_amdgcn_tensor_load_to_lds(g0, g1, z4, z4, z8, 0);
}
#else
#define USE_TDM 0
#endif

// ---------------------------------------------------------------------------
// fp32 -> bf16 elementwise
// ---------------------------------------------------------------------------
__global__ void cvt_bf16_kernel(const float* __restrict__ in,
                                uint16_t* __restrict__ out, int n) {
  int i = blockIdx.x * 256 + threadIdx.x;
  if (i < n) out[i] = f2bf(in[i]);
}

// ---------------------------------------------------------------------------
// RMSNorm: one block per row of D elements; output bf16
// ---------------------------------------------------------------------------
__global__ __launch_bounds__(256) void rmsnorm_kernel(
    const float* __restrict__ x, const float* __restrict__ w,
    uint16_t* __restrict__ out, int D) {
  const size_t row = blockIdx.x;
  const float* xr = x + row * (size_t)D;
  float ss = 0.f;
  for (int i = threadIdx.x; i < D; i += 256) { float v = xr[i]; ss += v * v; }
  #pragma unroll
  for (int d = 1; d < 32; d <<= 1) ss += __shfl_xor(ss, d, 32);
  __shared__ float red[8];
  if ((threadIdx.x & 31) == 0) red[threadIdx.x >> 5] = ss;
  __syncthreads();
  float tot = 0.f;
  #pragma unroll
  for (int j = 0; j < 8; ++j) tot += red[j];
  float inv = rsqrtf(tot / (float)D + 1e-6f);
  uint16_t* orow = out + row * (size_t)D;
  for (int i = threadIdx.x; i < D; i += 256) orow[i] = f2bf(xr[i] * inv * w[i]);
}

// ---------------------------------------------------------------------------
// Tiled bf16 WMMA GEMM: C[M,N] = A[M,K] @ W[N,K]^T  (both row-major over K)
// CTA 128x128, 8 waves each 64x32, K-step 32. MODE (compile-time):
//   0: Cf = acc                      (fp32 out)
//   1: Ch = bf16(silu(acc))         (bf16 out)
//   2: Ch = bf16(acc * bf2f(gmul))  (bf16 out)
//   3: Cf = acc + resid             (fp32 out)
// ---------------------------------------------------------------------------
#define GP 40   // LDS row pitch in bf16 elems (64B data + 16B pad = 80B)

template <int MODE>
__global__ __launch_bounds__(256) void gemm_bf16_kernel(
    const uint16_t* __restrict__ A, const uint16_t* __restrict__ W,
    float* __restrict__ Cf, uint16_t* __restrict__ Ch,
    const float* __restrict__ resid, const uint16_t* __restrict__ gmul,
    int M, int N, int K) {
  __shared__ uint16_t As[128 * GP];
  __shared__ uint16_t Bs[128 * GP];
  const int t = threadIdx.x;
  const int wave = t >> 5, lane = t & 31, half = lane >> 4, nlo = lane & 15;
  const int wM = (wave >> 2) * 64, wN = (wave & 3) * 32;
  const int gm0 = blockIdx.y * 128, gn0 = blockIdx.x * 128;

  v8f acc[4][2];
  #pragma unroll
  for (int m = 0; m < 4; ++m)
    #pragma unroll
    for (int n = 0; n < 2; ++n) acc[m][n] = vzero8();

#if USE_TDM
  const uint32_t asOff = (uint32_t)(uintptr_t)(void*)&As[0];
  const uint32_t bsOff = (uint32_t)(uintptr_t)(void*)&Bs[0];
#endif

  for (int k0 = 0; k0 < K; k0 += 32) {
#if USE_TDM
    if (wave == 0) {                        // one TDM issue per CTA (2 tiles)
      tdm_load_tile_128x32(A + (size_t)gm0 * K + k0, asOff, M, K, K);
      tdm_load_tile_128x32(W + (size_t)gn0 * K + k0, bsOff, N, K, K);
      __builtin_amdgcn_s_wait_tensorcnt(0);
    }
    __syncthreads();
#else
    #pragma unroll
    for (int it = 0; it < 2; ++it) {
      int idx = t + it * 256;
      int r = idx >> 2, c = (idx & 3) << 3;
      *(uint4*)&As[r * GP + c] = *(const uint4*)&A[(size_t)(gm0 + r) * K + k0 + c];
      *(uint4*)&Bs[r * GP + c] = *(const uint4*)&W[(size_t)(gn0 + r) * K + k0 + c];
      if (k0 + 32 < K) {                    // prefetch next k-tile
        __builtin_prefetch(&A[(size_t)(gm0 + r) * K + k0 + 32 + c], 0, 0);
        __builtin_prefetch(&W[(size_t)(gn0 + r) * K + k0 + 32 + c], 0, 0);
      }
    }
    __syncthreads();
#endif
    v16bf af[4], bfv[2];
    #pragma unroll
    for (int m = 0; m < 4; ++m) af[m] = frag_a(&As[(wM + m * 16) * GP], GP);
    #pragma unroll
    for (int n = 0; n < 2; ++n) bfv[n] = frag_b(&Bs[(wN + n * 16) * GP], GP);
    #pragma unroll
    for (int m = 0; m < 4; ++m)
      #pragma unroll
      for (int n = 0; n < 2; ++n)
        acc[m][n] = __builtin_amdgcn_wmma_f32_16x16x32_bf16(
            false, af[m], false, bfv[n], (short)0, acc[m][n], false, false);
    __syncthreads();
  }

  // epilogue: C layout -> VGPR i: row = i + 8*half, col = lane%16
  #pragma unroll
  for (int m = 0; m < 4; ++m)
    #pragma unroll
    for (int n = 0; n < 2; ++n)
      #pragma unroll
      for (int i = 0; i < 8; ++i) {
        int gr = gm0 + wM + m * 16 + i + half * 8;
        int gc = gn0 + wN + n * 16 + nlo;
        size_t o = (size_t)gr * N + gc;
        float v = acc[m][n][i];
        if (MODE == 0)      Cf[o] = v;
        else if (MODE == 1) Ch[o] = f2bf(v / (1.f + __expf(-v)));
        else if (MODE == 2) Ch[o] = f2bf(v * bf2f(gmul[o]));
        else                Cf[o] = v + resid[o];
      }
}

// ---------------------------------------------------------------------------
// RoPE for Q (fp32 in [B*S, H*HD] -> bf16 out, same layout)
// ---------------------------------------------------------------------------
__global__ void rope_q_kernel(const float* __restrict__ q,
                              uint16_t* __restrict__ qb) {
  int tid = blockIdx.x * 256 + threadIdx.x;   // B*S*H*64
  if (tid >= CM * CH * 64) return;
  int j = tid & 63;
  int h = (tid >> 6) & (CH - 1);
  int bs = tid >> 10;
  int s = bs & (CS - 1);
  float inv = __powf(10000.f, -(float)(2 * j) / 128.f);
  float ang = (float)s * inv;
  float c = __cosf(ang), sn = __sinf(ang);
  size_t base = (size_t)bs * CD + h * CHD;
  float t1 = q[base + j], t2 = q[base + 64 + j];
  qb[base + j]      = f2bf(t1 * c - t2 * sn);
  qb[base + 64 + j] = f2bf(t2 * c + t1 * sn);
}

// RoPE for K + bf16-cast for V (kv fp32 [B*S, 256]: first 128 = K, next = V)
__global__ void rope_kv_kernel(const float* __restrict__ kv,
                               uint16_t* __restrict__ kb,
                               uint16_t* __restrict__ vb) {
  int tid = blockIdx.x * 256 + threadIdx.x;   // B*S*64
  if (tid >= CM * 64) return;
  int j = tid & 63;
  int bs = tid >> 6;
  int s = bs & (CS - 1);
  float inv = __powf(10000.f, -(float)(2 * j) / 128.f);
  float ang = (float)s * inv;
  float c = __cosf(ang), sn = __sinf(ang);
  size_t ib = (size_t)bs * 256;
  size_t ob = (size_t)bs * CHD;
  float k1 = kv[ib + j], k2 = kv[ib + 64 + j];
  kb[ob + j]      = f2bf(k1 * c - k2 * sn);
  kb[ob + 64 + j] = f2bf(k2 * c + k1 * sn);
  vb[ob + j]      = f2bf(kv[ib + 128 + j]);
  vb[ob + 64 + j] = f2bf(kv[ib + 192 + j]);
}

// ---------------------------------------------------------------------------
// Flash attention with ALiBi + causal mask.
// Grid (S/64, H, B); 128 threads = 4 waves, wave owns 16 query rows.
// ---------------------------------------------------------------------------
#define KP 136   // K tile pitch (128+8)
#define VP 40    // V^T / P tile pitch (32+8)

__global__ __launch_bounds__(128) void attn_flash_kernel(
    const uint16_t* __restrict__ Qb, const uint16_t* __restrict__ Kb,
    const uint16_t* __restrict__ Vb, uint16_t* __restrict__ Ob) {
  __shared__ uint16_t Kt[32 * KP];        // K tile [key][hd]
  __shared__ uint16_t Vt[128 * VP];       // V tile transposed [hd][key]
  __shared__ uint16_t Pt[4 * 16 * VP];    // P per wave [row][key]
  const int t = threadIdx.x, wave = t >> 5, lane = t & 31;
  const int half = lane >> 4, nlo = lane & 15;
  const int q0 = blockIdx.x * 64, h = blockIdx.y, b = blockIdx.z;
  const float slope = exp2f(-0.5f * (float)(h + 1));   // 2^(-8(h+1)/H), H=16
  const float scale = 0.08838834764831845f;            // 1/sqrt(128)

  // Q fragments: 16 rows x 128 hd = 4 A-frags, loaded straight from global
  v16bf qa[4];
  {
    const uint16_t* qp =
        Qb + ((size_t)(b * CS + q0 + wave * 16 + nlo)) * CD + h * CHD;
    #pragma unroll
    for (int ks = 0; ks < 4; ++ks) {
      union { v16bf v; uint4 q[2]; } u;
      int c = ks * 32 + half * 8;
      u.q[0] = *(const uint4*)(qp + c);
      u.q[1] = *(const uint4*)(qp + c + 16);
      qa[ks] = u.v;
    }
  }
  v8f o[8];
  #pragma unroll
  for (int nh = 0; nh < 8; ++nh) o[nh] = vzero8();
  float mrun[8], lrun[8];
  #pragma unroll
  for (int i = 0; i < 8; ++i) { mrun[i] = -1e30f; lrun[i] = 0.f; }

  const int ktmax = (q0 + 63) >> 5;
  for (int kt = 0; kt <= ktmax; ++kt) {
    const int kb0 = kt * 32;
    // cooperative K/V tile load (V transposed into Vt)
    #pragma unroll
    for (int it = 0; it < 4; ++it) {
      int idx = t + it * 128;
      int r = idx >> 3, c = (idx & 7) << 4;
      size_t gro = ((size_t)(b * CS + kb0 + r)) * CHD + c;
      *(uint4*)&Kt[r * KP + c] = *(const uint4*)&Kb[gro];
      uint4 d = *(const uint4*)&Vb[gro];
      const uint16_t* ds = (const uint16_t*)&d;
      #pragma unroll
      for (int jj = 0; jj < 8; ++jj) Vt[(c + jj) * VP + r] = ds[jj];
    }
    __syncthreads();

    // scores = Q @ K^T : two 16-key subtiles x 4 k-steps over HD
    v8f sc[2]; sc[0] = vzero8(); sc[1] = vzero8();
    #pragma unroll
    for (int n = 0; n < 2; ++n)
      #pragma unroll
      for (int ks = 0; ks < 4; ++ks) {
        union { v16bf v; uint4 q[2]; } u;
        const uint16_t* p = &Kt[(n * 16 + nlo) * KP + ks * 32 + half * 8];
        u.q[0] = *(const uint4*)p;
        u.q[1] = *(const uint4*)(p + 16);
        sc[n] = __builtin_amdgcn_wmma_f32_16x16x32_bf16(
            false, qa[ks], false, u.v, (short)0, sc[n], false, false);
      }

    // scale + ALiBi + causal; online softmax (row = i + 8*half per C layout)
    float mx[8];
    #pragma unroll
    for (int i = 0; i < 8; ++i) {
      int qr = q0 + wave * 16 + i + half * 8;
      float best = -3e38f;
      #pragma unroll
      for (int n = 0; n < 2; ++n) {
        int kc = kb0 + n * 16 + nlo;
        float v = sc[n][i] * scale - slope * fabsf((float)(qr - kc));
        if (kc > qr) v = -1e9f;
        sc[n][i] = v;
        best = fmaxf(best, v);
      }
      mx[i] = best;
    }
    float corr[8];
    #pragma unroll
    for (int i = 0; i < 8; ++i) {
      #pragma unroll
      for (int d = 1; d < 16; d <<= 1)
        mx[i] = fmaxf(mx[i], __shfl_xor(mx[i], d, 32));
      float nm = fmaxf(mrun[i], mx[i]);
      corr[i] = __expf(mrun[i] - nm);
      mrun[i] = nm;
      float s = 0.f;
      #pragma unroll
      for (int n = 0; n < 2; ++n) {
        float p = __expf(sc[n][i] - nm);
        sc[n][i] = p;
        s += p;
      }
      #pragma unroll
      for (int d = 1; d < 16; d <<= 1) s += __shfl_xor(s, d, 32);
      lrun[i] = lrun[i] * corr[i] + s;
    }
    #pragma unroll
    for (int nh = 0; nh < 8; ++nh)
      #pragma unroll
      for (int i = 0; i < 8; ++i) o[nh][i] *= corr[i];

    // P (C layout) -> LDS -> reload in A layout
    uint16_t* pw = &Pt[wave * 16 * VP];
    #pragma unroll
    for (int n = 0; n < 2; ++n)
      #pragma unroll
      for (int i = 0; i < 8; ++i)
        pw[(i + half * 8) * VP + n * 16 + nlo] = f2bf(sc[n][i]);
    asm volatile("s_wait_dscnt 0" ::: "memory");   // intra-wave LDS RAW
    v16bf pa = frag_a(pw, VP);

    // O += P @ V : 8 hd-subtiles
    #pragma unroll
    for (int nh = 0; nh < 8; ++nh) {
      union { v16bf v; uint4 q[2]; } u;
      const uint16_t* p = &Vt[(nh * 16 + nlo) * VP + half * 8];
      u.q[0] = *(const uint4*)p;
      u.q[1] = *(const uint4*)(p + 16);
      o[nh] = __builtin_amdgcn_wmma_f32_16x16x32_bf16(
          false, pa, false, u.v, (short)0, o[nh], false, false);
    }
    __syncthreads();
  }

  // normalize + store bf16 to [B,S,D] (d = h*128 + hd)
  #pragma unroll
  for (int nh = 0; nh < 8; ++nh)
    #pragma unroll
    for (int i = 0; i < 8; ++i) {
      int qr = q0 + wave * 16 + i + half * 8;
      Ob[((size_t)(b * CS + qr)) * CD + h * CHD + nh * 16 + nlo] =
          f2bf(o[nh][i] / lrun[i]);
    }
}

// ---------------------------------------------------------------------------
// Host orchestration
// ---------------------------------------------------------------------------
extern "C" void kernel_launch(void* const* d_in, const int* in_sizes, int n_in,
                              void* d_out, int out_size, void* d_ws,
                              size_t ws_size, hipStream_t stream) {
  (void)in_sizes; (void)n_in; (void)out_size; (void)ws_size;
  const float* x   = (const float*)d_in[0];
  // d_in[1] = causal mask (recomputed inline)
  const float* wn1 = (const float*)d_in[2];
  const float* wn2 = (const float*)d_in[3];
  const float* qw  = (const float*)d_in[4];
  const float* kvw = (const float*)d_in[5];
  const float* ow  = (const float*)d_in[6];
  const float* w1  = (const float*)d_in[7];
  const float* w2  = (const float*)d_in[8];
  const float* w3  = (const float*)d_in[9];
  float* out = (float*)d_out;

  char* ws = (char*)d_ws;
  size_t off = 0;
  auto alloc = [&](size_t bytes) -> char* {
    char* p = ws + off;
    off += (bytes + 255) & ~(size_t)255;
    return p;
  };
  uint16_t* qw_b  = (uint16_t*)alloc((size_t)CD * CD * 2);
  uint16_t* kvw_b = (uint16_t*)alloc((size_t)256 * CD * 2);
  uint16_t* ow_b  = (uint16_t*)alloc((size_t)CD * CD * 2);
  uint16_t* w1_b  = (uint16_t*)alloc((size_t)CFF * CD * 2);
  uint16_t* w2_b  = (uint16_t*)alloc((size_t)CFF * CD * 2);
  uint16_t* w3_b  = (uint16_t*)alloc((size_t)CFF * CD * 2);
  uint16_t* h_b   = (uint16_t*)alloc((size_t)CM * CD * 2);   // h, reused as h2
  float*    x2    = (float*)   alloc((size_t)CM * CD * 4);
  char* regionA   = alloc(73400320);   // q_f32,kv_f32,q_b,k_b,v_b,attn_b; reused as g
  uint16_t* gated = (uint16_t*)alloc((size_t)CM * CFF * 2);

  float*    qf  = (float*)regionA;
  float*    kvf = (float*)(regionA + (size_t)CM * CD * 4);
  uint16_t* q_b = (uint16_t*)(regionA + (size_t)CM * CD * 4 + (size_t)CM * 256 * 4);
  uint16_t* k_b = q_b + (size_t)CM * CD;
  uint16_t* v_b = k_b + (size_t)CM * CHD;
  uint16_t* at_b = v_b + (size_t)CM * CHD;
  uint16_t* g_b  = (uint16_t*)regionA;   // silu gate, after region A is free

  auto cvt = [&](const float* src, uint16_t* dst, size_t n) {
    cvt_bf16_kernel<<<dim3((unsigned)((n + 255) / 256)), 256, 0, stream>>>(
        src, dst, (int)n);
  };
  cvt(qw,  qw_b,  (size_t)CD * CD);
  cvt(kvw, kvw_b, (size_t)256 * CD);
  cvt(ow,  ow_b,  (size_t)CD * CD);
  cvt(w1,  w1_b,  (size_t)CFF * CD);
  cvt(w2,  w2_b,  (size_t)CFF * CD);
  cvt(w3,  w3_b,  (size_t)CFF * CD);

  // h = rmsnorm(x) (bf16)
  rmsnorm_kernel<<<CM, 256, 0, stream>>>(x, wn1, h_b, CD);
  // q = h @ qw^T (fp32) ; kv = h @ kvw^T (fp32)
  gemm_bf16_kernel<0><<<dim3(CD/128, CM/128), 256, 0, stream>>>(
      h_b, qw_b, qf, nullptr, nullptr, nullptr, CM, CD, CD);
  gemm_bf16_kernel<0><<<dim3(256/128, CM/128), 256, 0, stream>>>(
      h_b, kvw_b, kvf, nullptr, nullptr, nullptr, CM, 256, CD);
  // RoPE
  rope_q_kernel<<<(CM * CH * 64) / 256, 256, 0, stream>>>(qf, q_b);
  rope_kv_kernel<<<(CM * 64) / 256, 256, 0, stream>>>(kvf, k_b, v_b);
  // flash attention -> bf16 [B,S,D]
  attn_flash_kernel<<<dim3(CS/64, CH, CB), 128, 0, stream>>>(q_b, k_b, v_b, at_b);
  // x2 = x + attn @ ow^T (fp32)
  gemm_bf16_kernel<3><<<dim3(CD/128, CM/128), 256, 0, stream>>>(
      at_b, ow_b, x2, nullptr, x, nullptr, CM, CD, CD);
  // h2 = rmsnorm(x2) (bf16, reuses h slot)
  rmsnorm_kernel<<<CM, 256, 0, stream>>>(x2, wn2, h_b, CD);
  // g = silu(h2 @ w1^T) (bf16) ; gated = (h2 @ w2^T) * g (bf16)
  gemm_bf16_kernel<1><<<dim3(CFF/128, CM/128), 256, 0, stream>>>(
      h_b, w1_b, nullptr, g_b, nullptr, nullptr, CM, CFF, CD);
  gemm_bf16_kernel<2><<<dim3(CFF/128, CM/128), 256, 0, stream>>>(
      h_b, w2_b, nullptr, gated, nullptr, g_b, CM, CFF, CD);
  // out = x2 + gated @ w3^T (fp32)
  gemm_bf16_kernel<3><<<dim3(CD/128, CM/128), 256, 0, stream>>>(
      gated, w3_b, out, nullptr, x2, nullptr, CM, CD, CFF);
}
